// SelfAttentionLayer_16045997818211
// MI455X (gfx1250) — matrix operand order repaired and checked by reading, not compiled, not run on hardware
//
#include <hip/hip_runtime.h>
#include <hip/hip_bf16.h>
#include <math.h>
#include <stdint.h>

typedef __attribute__((ext_vector_type(16))) __bf16 v16bf;
typedef __attribute__((ext_vector_type(8)))  float  v8f;

#define N_TOK  2048
#define DMODEL 1024
#define NHEAD  16
#define HD     64

static __device__ __forceinline__ unsigned short f2bf(float f) {
  union { float f; unsigned u; } v; v.f = f;
  unsigned u = v.u;
  unsigned r = u + 0x7FFFu + ((u >> 16) & 1u);   // round-to-nearest-even
  return (unsigned short)(r >> 16);
}

union Frag { v16bf v; uint4 q[2]; };

static __device__ __forceinline__ v8f wmma_bf16(const Frag a, const Frag b, v8f c) {
  return __builtin_amdgcn_wmma_f32_16x16x32_bf16(false, a.v, false, b.v, (short)0, c, false, false);
}

// async DMA: global -> LDS, 16 bytes per lane, tracked by ASYNCcnt (no VGPR staging)
static __device__ __forceinline__ void async_b128(unsigned lds_byte_off, const void* gptr) {
  asm volatile("global_load_async_to_lds_b128 %0, %1, off"
               :: "v"(lds_byte_off), "v"(gptr) : "memory");
}
#define WAIT_ASYNC(n) asm volatile("s_wait_asynccnt " #n ::: "memory")

// LDS flat address: low 32 bits of the flat pointer are the LDS byte offset
static __device__ __forceinline__ unsigned lds_off(const void* p) {
  return (unsigned)(uintptr_t)p;
}

// ---------------- fp32 -> bf16 conversion ----------------
__global__ __launch_bounds__(256) void cvt_kernel(const float* __restrict__ in,
                                                  unsigned short* __restrict__ out, int n) {
  int i = (blockIdx.x * 256 + threadIdx.x) * 4;
  if (i + 3 < n) {
    float4 v = *(const float4*)&in[i];
    out[i + 0] = f2bf(v.x);
    out[i + 1] = f2bf(v.y);
    out[i + 2] = f2bf(v.z);
    out[i + 3] = f2bf(v.w);
  }
}

// fp32 W[k][n] (1024x1024) -> bf16 WT[n][k] so GEMM B-fragments read contiguous K
__global__ __launch_bounds__(256) void cvtT_kernel(const float* __restrict__ in,
                                                   unsigned short* __restrict__ out) {
  int idx = blockIdx.x * 256 + threadIdx.x;
  int k = idx >> 10, n = idx & 1023;
  out[(size_t)n * DMODEL + k] = f2bf(in[idx]);
}

// ---------------- bf16 GEMM: C(2048x1024) = A @ BT^T + bias [+resid] ----------------
// Double-buffered LDS tiles filled by async-to-LDS DMA, overlapped with WMMA.
// mode 0: Cb (bf16) = acc + bias      mode 1: Cf (fp32) = acc + bias + resid
__global__ __launch_bounds__(256) void gemm_bf16_kernel(
    const unsigned short* __restrict__ A,    // M x K bf16 row-major
    const unsigned short* __restrict__ BT,   // N x K bf16 (B transposed)
    const float* __restrict__ bias,
    unsigned short* __restrict__ Cb,
    float* __restrict__ Cf,
    const float* __restrict__ resid,
    int mode) {
  const int K = DMODEL;
  __shared__ unsigned short As[2][128 * 40];  // rows padded 32->40 (bank spread, 16B align)
  __shared__ unsigned short Bs[2][128 * 40];
  const int t = threadIdx.x;
  const int lane = t & 31, w = t >> 5;
  const int lh = lane >> 4, ll = lane & 15;
  const int wr = w & 1, wc = w >> 1;          // 2x4 wave grid, each wave: 64x32 of the 128x128 tile
  const int row0 = blockIdx.y * 128, col0 = blockIdx.x * 128;

  // per-thread DMA slots: chunks t and t+256 of 512 (128 rows x 4 uint4 chunks)
  const int fr = t >> 2, fch = (t & 3) << 3;  // rows fr and fr+64, 8 bf16 per chunk
  const unsigned ldsA = lds_off(&As[0][0]);
  const unsigned ldsB = lds_off(&Bs[0][0]);
  const unsigned bufBytes = 128 * 40 * 2;
  const unsigned o1 = (unsigned)((fr * 40 + fch) * 2);
  const unsigned o2 = (unsigned)(((fr + 64) * 40 + fch) * 2);
  const unsigned short* gA1 = A  + (size_t)(row0 + fr) * K + fch;
  const unsigned short* gA2 = A  + (size_t)(row0 + fr + 64) * K + fch;
  const unsigned short* gB1 = BT + (size_t)(col0 + fr) * K + fch;
  const unsigned short* gB2 = BT + (size_t)(col0 + fr + 64) * K + fch;

  v8f acc[4][2];
  v8f zero = {0.f, 0.f, 0.f, 0.f, 0.f, 0.f, 0.f, 0.f};
#pragma unroll
  for (int rt = 0; rt < 4; ++rt)
#pragma unroll
    for (int ct = 0; ct < 2; ++ct) acc[rt][ct] = zero;

  // prologue: stage 0 into buffer 0
  async_b128(ldsA + o1, gA1);
  async_b128(ldsA + o2, gA2);
  async_b128(ldsB + o1, gB1);
  async_b128(ldsB + o2, gB2);

  int buf = 0;
  for (int kb = 0; kb < K; kb += 32) {
    if (kb + 32 < K) {                        // issue next stage into the other buffer
      unsigned nb = (unsigned)(buf ^ 1) * bufBytes;
      async_b128(ldsA + nb + o1, gA1 + kb + 32);
      async_b128(ldsA + nb + o2, gA2 + kb + 32);
      async_b128(ldsB + nb + o1, gB1 + kb + 32);
      async_b128(ldsB + nb + o2, gB2 + kb + 32);
      WAIT_ASYNC(0x4);                        // current stage done; next stage in flight
    } else {
      WAIT_ASYNC(0x0);
    }
    __syncthreads();

    Frag a[4], b[2];
#pragma unroll
    for (int rt = 0; rt < 4; ++rt) {
      const unsigned short* p = &As[buf][(wr * 64 + rt * 16 + ll) * 40 + lh * 8];
      a[rt].q[0] = *(const uint4*)p;          // K = kbase..kbase+7
      a[rt].q[1] = *(const uint4*)(p + 16);   // K = 16+kbase..16+kbase+7
    }
#pragma unroll
    for (int ct = 0; ct < 2; ++ct) {
      const unsigned short* p = &Bs[buf][(wc * 32 + ct * 16 + ll) * 40 + lh * 8];
      b[ct].q[0] = *(const uint4*)p;
      b[ct].q[1] = *(const uint4*)(p + 16);
    }
#pragma unroll
    for (int rt = 0; rt < 4; ++rt)
#pragma unroll
      for (int ct = 0; ct < 2; ++ct) acc[rt][ct] = wmma_bf16(a[rt], b[ct], acc[rt][ct]);

    __syncthreads();                          // reads done before this buffer is refilled
    buf ^= 1;
  }

#pragma unroll
  for (int rt = 0; rt < 4; ++rt)
#pragma unroll
    for (int ct = 0; ct < 2; ++ct)
#pragma unroll
      for (int i = 0; i < 8; ++i) {
        int row = row0 + wr * 64 + rt * 16 + i + 8 * lh;
        int col = col0 + wc * 32 + ct * 16 + ll;
        float vout = acc[rt][ct][i] + bias[col];
        size_t idx = (size_t)row * DMODEL + col;
        if (mode == 0) Cb[idx] = f2bf(vout);
        else           Cf[idx] = vout + resid[idx];
      }
}

// ---------------- flash-style multi-head attention (bf16 WMMA, fp32 softmax state) ----------------
__global__ __launch_bounds__(256) void attn_kernel(
    const unsigned short* __restrict__ Q,
    const unsigned short* __restrict__ K,
    const unsigned short* __restrict__ V,
    unsigned short* __restrict__ Ob) {          // scrambled bf16 output, 2048x1024
  __shared__ unsigned short Qs[128 * 72];       // [qrow][hd], padded
  __shared__ unsigned short Ks[64 * 72];        // [key][hd]
  __shared__ unsigned short Vt[64 * 72];        // [hd][key]  (transposed for contiguous B-frags)
  __shared__ unsigned short Ps[8 * 16 * 72];    // per-wave P staging (C-layout -> A-layout)

  const int t = threadIdx.x;
  const int lane = t & 31, w = t >> 5;
  const int lh = lane >> 4, ll = lane & 15;
  const int q0 = blockIdx.x * 128;
  const int head = blockIdx.y;
  const size_t hoff = (size_t)head * HD;

  // async DMA of the Q block (128x64): 4 chunks per thread
  {
    int r = t >> 3, ch = (t & 7) << 3;
    unsigned base = lds_off(&Qs[0]);
#pragma unroll
    for (int j = 0; j < 4; ++j) {
      async_b128(base + (unsigned)(((r + j * 32) * 72 + ch) * 2),
                 Q + (size_t)(q0 + r + j * 32) * DMODEL + hoff + ch);
    }
  }
  WAIT_ASYNC(0x0);
  __syncthreads();

  Frag aQ[2];                                   // wave's 16 query rows, K(=hd) split 0..31 / 32..63
#pragma unroll
  for (int ks = 0; ks < 2; ++ks) {
    const unsigned short* p = &Qs[(w * 16 + ll) * 72 + ks * 32 + lh * 8];
    aQ[ks].q[0] = *(const uint4*)p;
    aQ[ks].q[1] = *(const uint4*)(p + 16);
  }

  v8f zero = {0.f, 0.f, 0.f, 0.f, 0.f, 0.f, 0.f, 0.f};
  v8f o[4];
#pragma unroll
  for (int ht = 0; ht < 4; ++ht) o[ht] = zero;
  float m[8], l[8];
#pragma unroll
  for (int i = 0; i < 8; ++i) { m[i] = -__builtin_inff(); l[i] = 0.f; }

  const int fr = t >> 3, fch = (t & 7) << 3;    // K/V tile chunks: rows fr, fr+32
  const unsigned ldsK = lds_off(&Ks[0]);

  for (int kb = 0; kb < N_TOK; kb += 64) {
    __syncthreads();                            // prev-iteration reads of Ks/Vt done
    // K tile: async DMA straight to LDS
    async_b128(ldsK + (unsigned)((fr * 72 + fch) * 2),
               K + (size_t)(kb + fr) * DMODEL + hoff + fch);
    async_b128(ldsK + (unsigned)(((fr + 32) * 72 + fch) * 2),
               K + (size_t)(kb + fr + 32) * DMODEL + hoff + fch);
    // V tile: VGPR path (transposed on the way into LDS)
#pragma unroll
    for (int half = 0; half < 2; ++half) {
      int r = fr + half * 32;
      uint4 vv = *(const uint4*)&V[(size_t)(kb + r) * DMODEL + hoff + fch];
      const unsigned short* vs = (const unsigned short*)&vv;
#pragma unroll
      for (int j = 0; j < 8; ++j) Vt[(fch + j) * 72 + r] = vs[j];
    }
    WAIT_ASYNC(0x0);
    __syncthreads();

    // S = Q @ K^T  (16 rows x 64 keys per wave; 4 col-tiles x 2 k-steps)
    v8f s[4];
#pragma unroll
    for (int ct = 0; ct < 4; ++ct) s[ct] = zero;
#pragma unroll
    for (int ct = 0; ct < 4; ++ct)
#pragma unroll
      for (int ks = 0; ks < 2; ++ks) {
        Frag bk;
        const unsigned short* p = &Ks[(ct * 16 + ll) * 72 + ks * 32 + lh * 8];
        bk.q[0] = *(const uint4*)p;
        bk.q[1] = *(const uint4*)(p + 16);
        s[ct] = wmma_bf16(aQ[ks], bk, s[ct]);
      }

    // diagonal self-exclusion mask
#pragma unroll
    for (int ct = 0; ct < 4; ++ct) {
      int keyg = kb + ct * 16 + ll;
#pragma unroll
      for (int i = 0; i < 8; ++i)
        if (keyg == q0 + w * 16 + i + 8 * lh) s[ct][i] = -__builtin_inff();
    }

    // online softmax: row reductions across the 16 lanes of each half-wave
    float sc[8];
#pragma unroll
    for (int i = 0; i < 8; ++i) {
      float v0 = fmaxf(fmaxf(s[0][i], s[1][i]), fmaxf(s[2][i], s[3][i]));
      v0 = fmaxf(v0, __shfl_xor(v0, 1, 32));
      v0 = fmaxf(v0, __shfl_xor(v0, 2, 32));
      v0 = fmaxf(v0, __shfl_xor(v0, 4, 32));
      v0 = fmaxf(v0, __shfl_xor(v0, 8, 32));
      float mn = fmaxf(m[i], v0);
      sc[i] = __expf(m[i] - mn);
      m[i] = mn;
    }
#pragma unroll
    for (int ct = 0; ct < 4; ++ct)
#pragma unroll
      for (int i = 0; i < 8; ++i) s[ct][i] = __expf(s[ct][i] - m[i]);
#pragma unroll
    for (int i = 0; i < 8; ++i) {
      float rs = s[0][i] + s[1][i] + s[2][i] + s[3][i];
      rs += __shfl_xor(rs, 1, 32);
      rs += __shfl_xor(rs, 2, 32);
      rs += __shfl_xor(rs, 4, 32);
      rs += __shfl_xor(rs, 8, 32);
      l[i] = l[i] * sc[i] + rs;
    }
#pragma unroll
    for (int ht = 0; ht < 4; ++ht)
#pragma unroll
      for (int i = 0; i < 8; ++i) o[ht][i] *= sc[i];

    // stage P (C-layout regs) -> LDS -> A-fragment layout
#pragma unroll
    for (int ct = 0; ct < 4; ++ct)
#pragma unroll
      for (int i = 0; i < 8; ++i)
        Ps[(w * 16 + i + 8 * lh) * 72 + ct * 16 + ll] = f2bf(s[ct][i]);
    __syncthreads();

    Frag aP[2];
#pragma unroll
    for (int ks = 0; ks < 2; ++ks) {
      const unsigned short* p = &Ps[(w * 16 + ll) * 72 + ks * 32 + lh * 8];
      aP[ks].q[0] = *(const uint4*)p;
      aP[ks].q[1] = *(const uint4*)(p + 16);
    }
    // O += P @ V  (4 hd-tiles x 2 key-steps)
#pragma unroll
    for (int ht = 0; ht < 4; ++ht)
#pragma unroll
      for (int ks = 0; ks < 2; ++ks) {
        Frag bv;
        const unsigned short* p = &Vt[(ht * 16 + ll) * 72 + ks * 32 + lh * 8];
        bv.q[0] = *(const uint4*)p;
        bv.q[1] = *(const uint4*)(p + 16);
        o[ht] = wmma_bf16(aP[ks], bv, o[ht]);
      }
  }

  // finalize + store with the reference's transpose(1,2).view(N,D) scramble:
  // O.flat[(head*64+hd)*2048 + q] -> row=(head*64+hd)*2 + q/1024, col=q%1024
#pragma unroll
  for (int ht = 0; ht < 4; ++ht)
#pragma unroll
    for (int i = 0; i < 8; ++i) {
      int qg = q0 + w * 16 + i + 8 * lh;
      int hd = ht * 16 + ll;
      float vout = o[ht][i] / l[i];
      int r = (head * HD + hd) * 2 + (qg >> 10);
      int c2 = qg & 1023;
      Ob[(size_t)r * DMODEL + c2] = f2bf(vout);
    }
}

// ---------------- LayerNorm: one row (1024) per block ----------------
__global__ __launch_bounds__(256) void ln_kernel(const float* __restrict__ Y,
                                                 const float* __restrict__ g,
                                                 const float* __restrict__ b,
                                                 float* __restrict__ out) {
  const int row = blockIdx.x;
  const int t = threadIdx.x;
  const int lane = t & 31, w = t >> 5;
  __shared__ float s1[8], s2[8];
  float4 y = *(const float4*)&Y[(size_t)row * DMODEL + t * 4];
  float a = y.x + y.y + y.z + y.w;
  float q = y.x * y.x + y.y * y.y + y.z * y.z + y.w * y.w;
#pragma unroll
  for (int msk = 1; msk < 32; msk <<= 1) {
    a += __shfl_xor(a, msk, 32);
    q += __shfl_xor(q, msk, 32);
  }
  if (lane == 0) { s1[w] = a; s2[w] = q; }
  __syncthreads();
  float S = 0.f, S2 = 0.f;
#pragma unroll
  for (int j = 0; j < 8; ++j) { S += s1[j]; S2 += s2[j]; }
  float mu  = S * (1.f / DMODEL);
  float var = S2 * (1.f / DMODEL) - mu * mu;
  float inv = rsqrtf(var + 1e-5f);
  float4 gg = *(const float4*)&g[t * 4];
  float4 bb = *(const float4*)&b[t * 4];
  float4 r;
  r.x = (y.x - mu) * inv * gg.x + bb.x;
  r.y = (y.y - mu) * inv * gg.y + bb.y;
  r.z = (y.z - mu) * inv * gg.z + bb.z;
  r.w = (y.w - mu) * inv * gg.w + bb.w;
  *(float4*)&out[(size_t)row * DMODEL + t * 4] = r;
}

// ---------------- host orchestration ----------------
extern "C" void kernel_launch(void* const* d_in, const int* in_sizes, int n_in,
                              void* d_out, int out_size, void* d_ws, size_t ws_size,
                              hipStream_t stream) {
  (void)in_sizes; (void)n_in; (void)out_size; (void)ws_size;
  const float* X  = (const float*)d_in[0];
  const float* Wq = (const float*)d_in[1];
  const float* bq = (const float*)d_in[2];
  const float* Wk = (const float*)d_in[3];
  const float* bk = (const float*)d_in[4];
  const float* Wv = (const float*)d_in[5];
  const float* bv = (const float*)d_in[6];
  const float* Wo = (const float*)d_in[7];
  const float* bo = (const float*)d_in[8];
  const float* g  = (const float*)d_in[9];
  const float* b  = (const float*)d_in[10];

  char* ws = (char*)d_ws;
  const size_t ND2 = (size_t)N_TOK * DMODEL * 2;   // 4 MB (bf16 2048x1024)
  const size_t DD2 = (size_t)DMODEL * DMODEL * 2;  // 2 MB (bf16 1024x1024)
  unsigned short* Xb   = (unsigned short*)(ws);
  unsigned short* WqT  = (unsigned short*)(ws + ND2);
  unsigned short* WkT  = (unsigned short*)(ws + ND2 + 1 * DD2);
  unsigned short* WvT  = (unsigned short*)(ws + ND2 + 2 * DD2);
  unsigned short* WoT  = (unsigned short*)(ws + ND2 + 3 * DD2);
  unsigned short* Qb   = (unsigned short*)(ws + 1 * ND2 + 4 * DD2);
  unsigned short* Kb   = (unsigned short*)(ws + 2 * ND2 + 4 * DD2);
  unsigned short* Vb   = (unsigned short*)(ws + 3 * ND2 + 4 * DD2);
  unsigned short* Obuf = (unsigned short*)(ws + 4 * ND2 + 4 * DD2);
  float*          Y    = (float*)         (ws + 5 * ND2 + 4 * DD2);

  cvt_kernel <<<2048, 256, 0, stream>>>(X, Xb, N_TOK * DMODEL);
  cvtT_kernel<<<4096, 256, 0, stream>>>(Wq, WqT);
  cvtT_kernel<<<4096, 256, 0, stream>>>(Wk, WkT);
  cvtT_kernel<<<4096, 256, 0, stream>>>(Wv, WvT);
  cvtT_kernel<<<4096, 256, 0, stream>>>(Wo, WoT);

  dim3 ggrid(DMODEL / 128, N_TOK / 128);
  gemm_bf16_kernel<<<ggrid, 256, 0, stream>>>(Xb, WqT, bq, Qb, nullptr, nullptr, 0);
  gemm_bf16_kernel<<<ggrid, 256, 0, stream>>>(Xb, WkT, bk, Kb, nullptr, nullptr, 0);
  gemm_bf16_kernel<<<ggrid, 256, 0, stream>>>(Xb, WvT, bv, Vb, nullptr, nullptr, 0);

  attn_kernel<<<dim3(N_TOK / 128, NHEAD), 256, 0, stream>>>(Qb, Kb, Vb, Obuf);

  gemm_bf16_kernel<<<ggrid, 256, 0, stream>>>(Obuf, WoT, bo, nullptr, Y, X, 1);

  ln_kernel<<<N_TOK, 256, 0, stream>>>(Y, g, b, (float*)d_out);
}
